// GCN_83889301225632
// MI455X (gfx1250) — matrix operand order repaired and checked by reading, not compiled, not run on hardware
//
#include <hip/hip_runtime.h>
#include <hip/hip_bf16.h>

// ---------------------------------------------------------------------------
// GNN forward for MI455X (gfx1250).
// Dominant cost: two NNConv layers, each rewritten as a fused GEMM
//   [E, 1024] x [1024, 32]  with A-rows = outer(h_src[e], ea[e]) built in
// registers, B = reshaped W*_nn resident in LDS (staged via the Tensor Data
// Mover when available), computed with v_wmma_f32_16x16x32_f16 (f16 operands,
// f32 accumulate), scatter-add via global_atomic_add_f32.
// ---------------------------------------------------------------------------

typedef __attribute__((ext_vector_type(16))) _Float16 v16h;
typedef __attribute__((ext_vector_type(8)))  _Float16 v8h;
typedef __attribute__((ext_vector_type(8)))  float    v8f;
typedef __attribute__((ext_vector_type(4)))  unsigned int v4u;
typedef __attribute__((ext_vector_type(8)))  int      v8i;
typedef __attribute__((ext_vector_type(4)))  int      v4i;

#if defined(__has_builtin)
#if __has_builtin(__builtin_amdgcn_tensor_load_to_lds) && \
    __has_builtin(__builtin_amdgcn_s_wait_tensorcnt)
#define USE_TDM 1
#endif
#endif

namespace {
enum : int { N = 25000, E = 100000, FN = 32, FE = 16, DH = 32,
             NH = 4, DG = 8, NC = 10, E2 = E + N };
}

// -------------------------- small helper kernels ---------------------------

__global__ void k_encode_nodes(const float* __restrict__ x,
                               const float* __restrict__ W,
                               const float* __restrict__ b,
                               float* __restrict__ h0,
                               _Float16* __restrict__ h0h) {
  int t = blockIdx.x * blockDim.x + threadIdx.x;
  if (t >= N * DH) return;
  int n = t >> 5, o = t & 31;
  float acc = b[o];
  #pragma unroll
  for (int i = 0; i < FN; ++i) acc += x[(size_t)n * FN + i] * W[i * DH + o];
  h0[t] = acc;
  h0h[t] = (_Float16)acc;
}

__global__ void k_encode_edges(const float* __restrict__ ea,
                               const float* __restrict__ W,
                               const float* __restrict__ b,
                               _Float16* __restrict__ eah) {
  int t = blockIdx.x * blockDim.x + threadIdx.x;
  if (t >= E * DH) return;
  int e = t >> 5, o = t & 31;
  float acc = b[o];
  #pragma unroll
  for (int i = 0; i < FE; ++i) acc += ea[(size_t)e * FE + i] * W[i * DH + o];
  eah[t] = (_Float16)acc;
}

// Pack W_nn [32, 1024] (indexed [kloc, i*32 + n]) into WMMA-B friendly layout:
// Wp[i][t][lane][j]: lane<16 -> column n = t*16 + lane, kloc = j;
//                    lane>=16 -> column n = t*16 + lane-16, kloc = 16 + j.
__global__ void k_pack_w(const float* __restrict__ Wnn, _Float16* __restrict__ Wp) {
  int idx = blockIdx.x * blockDim.x + threadIdx.x;
  if (idx >= 32 * 2 * 32 * 16) return;
  int j = idx & 15;
  int L = (idx >> 4) & 31;
  int t = (idx >> 9) & 1;
  int i = idx >> 10;
  int n    = t * 16 + (L & 15);
  int kloc = (L < 16) ? j : (16 + j);
  Wp[idx] = (_Float16)Wnn[(size_t)kloc * (DH * DH) + i * DH + n];
}

// hb[n,o] = sum_i h[n,i] * bvec[i*32+o]   (per-node bias term of NNConv)
__global__ void k_hb(const float* __restrict__ h, const float* __restrict__ bvec,
                     float* __restrict__ hb) {
  int t = blockIdx.x * blockDim.x + threadIdx.x;
  if (t >= N * DH) return;
  int n = t >> 5, o = t & 31;
  float acc = 0.f;
  #pragma unroll
  for (int i = 0; i < DH; ++i) acc += h[(size_t)n * DH + i] * bvec[i * DH + o];
  hb[t] = acc;
}

// h_out = relu(agg + h_in @ root + bias); optional f16 copy
__global__ void k_node_update(const float* __restrict__ agg,
                              const float* __restrict__ h_in,
                              const float* __restrict__ root,
                              const float* __restrict__ bias,
                              float* __restrict__ h_out,
                              _Float16* __restrict__ h_out_h) {
  int t = blockIdx.x * blockDim.x + threadIdx.x;
  if (t >= N * DH) return;
  int n = t >> 5, o = t & 31;
  float acc = agg[t] + bias[o];
  #pragma unroll
  for (int i = 0; i < DH; ++i) acc += h_in[(size_t)n * DH + i] * root[i * DH + o];
  acc = acc > 0.f ? acc : 0.f;
  h_out[t] = acc;
  if (h_out_h) h_out_h[t] = (_Float16)acc;
}

__global__ void k_xlxr(const float* __restrict__ h1,
                       const float* __restrict__ Wl, const float* __restrict__ bl,
                       const float* __restrict__ Wr, const float* __restrict__ br,
                       float* __restrict__ xl, float* __restrict__ xr) {
  int t = blockIdx.x * blockDim.x + threadIdx.x;
  if (t >= N * DH) return;
  int n = t >> 5, o = t & 31;
  float al = bl[o], ar = br[o];
  #pragma unroll
  for (int i = 0; i < DH; ++i) {
    float hv = h1[(size_t)n * DH + i];
    al += hv * Wl[i * DH + o];
    ar += hv * Wr[i * DH + o];
  }
  xl[t] = al;
  xr[t] = ar;
}

__device__ __forceinline__ void atomicMaxFloat(float* addr, float val) {
  if (val >= 0.f) atomicMax((int*)addr, __float_as_int(val));
  else            atomicMin((unsigned int*)addr, __float_as_uint(val));
}

__global__ void k_init_minf(unsigned int* __restrict__ m, int count) {
  int t = blockIdx.x * blockDim.x + threadIdx.x;
  if (t < count) m[t] = 0xFF800000u;  // -inf
}

__global__ void k_gat_logits(const float* __restrict__ xl, const float* __restrict__ xr,
                             const float* __restrict__ att,
                             const int* __restrict__ src, const int* __restrict__ dst,
                             float* __restrict__ logits, float* __restrict__ m) {
  int t = blockIdx.x * blockDim.x + threadIdx.x;
  if (t >= E2 * NH) return;
  int e = t >> 2, h = t & 3;
  int si, di;
  if (e < E) { si = src[e]; di = dst[e]; } else { si = di = e - E; }
  float acc = 0.f;
  #pragma unroll
  for (int d = 0; d < DG; ++d) {
    float v = xl[(size_t)si * DH + h * DG + d] + xr[(size_t)di * DH + h * DG + d];
    v = v > 0.f ? v : 0.2f * v;  // leaky_relu(0.2)
    acc += v * att[h * DG + d];
  }
  logits[t] = acc;
  atomicMaxFloat(&m[di * NH + h], acc);
}

__global__ void k_gat_exp(float* __restrict__ logits, const float* __restrict__ m,
                          const int* __restrict__ dst, float* __restrict__ denom) {
  int t = blockIdx.x * blockDim.x + threadIdx.x;
  if (t >= E2 * NH) return;
  int e = t >> 2, h = t & 3;
  int di = (e < E) ? dst[e] : (e - E);
  float a = expf(logits[t] - m[di * NH + h]);
  logits[t] = a;
  atomicAdd(&denom[di * NH + h], a);
}

__global__ void k_gat_agg(const float* __restrict__ logits, const float* __restrict__ denom,
                          const float* __restrict__ xl,
                          const int* __restrict__ src, const int* __restrict__ dst,
                          float* __restrict__ outgat) {
  int t = blockIdx.x * blockDim.x + threadIdx.x;
  if (t >= E2 * NH) return;
  int e = t >> 2, h = t & 3;
  int si, di;
  if (e < E) { si = src[e]; di = dst[e]; } else { si = di = e - E; }
  float alpha = logits[t] / (denom[di * NH + h] + 1e-16f);
  #pragma unroll
  for (int d = 0; d < DG; ++d)
    atomicAdd(&outgat[(size_t)di * DH + h * DG + d],
              xl[(size_t)si * DH + h * DG + d] * alpha);
}

__global__ void k_gat_final(const float* __restrict__ outgat, const float* __restrict__ bias,
                            float* __restrict__ h2, _Float16* __restrict__ h2h) {
  int t = blockIdx.x * blockDim.x + threadIdx.x;
  if (t >= N * DH) return;
  int o = t & 31;
  float v = outgat[t] + bias[o];
  v = v > 0.f ? v : 0.f;
  h2[t] = v;
  h2h[t] = (_Float16)v;
}

__global__ void k_classify(const float* __restrict__ h3, const float* __restrict__ W,
                           const float* __restrict__ b, float* __restrict__ out) {
  int t = blockIdx.x * blockDim.x + threadIdx.x;
  if (t >= N * NC) return;
  int n = t / NC, c = t - n * NC;
  float acc = b[c];
  #pragma unroll
  for (int i = 0; i < DH; ++i) acc += h3[(size_t)n * DH + i] * W[i * NC + c];
  out[t] = acc;
}

// ------------------------- fused NNConv WMMA kernel ------------------------
// msg[e,o] = sum_{i,k} h[src[e],i]*ea[e,k]*Wnn[k,i*32+o]  (+ hb[src[e],o])
// GEMM: M = 16 edges/tile, K = 1024 (32 chunks of 32), Nout = 32 (2 tiles).
// A rows built in registers (outer product); B staged into LDS by the TDM.
// Requires E % 16 == 0 (E = 100000 = 6250*16).
__global__ __launch_bounds__(256) void k_nnconv_wmma(
    const _Float16* __restrict__ eah,   // [E,32] f16 encoded edges
    const _Float16* __restrict__ hh,    // [N,32] f16 node features
    const int* __restrict__ src, const int* __restrict__ dst,
    const _Float16* __restrict__ Wp,    // packed B: [32][2][32 lanes][16]
    const float* __restrict__ hb,       // [N,32] node-side bias term
    float* __restrict__ agg)            // [N,32] zero-initialized accumulator
{
  __shared__ __align__(16) _Float16 sW[32 * 2 * 32 * 16];  // 64 KB

#if defined(USE_TDM)
  // Tensor Data Mover: DMA the 64 KB packed-B block straight into LDS.
  // sW is the only LDS allocation in this kernel -> LDS offset 0.
  // D# per CDNA5 ISA: 1D copy expressed as a 2D tile of
  // 8192 x 1 elements with data_size = 8 bytes.
  if (threadIdx.x < 32) {   // one wave issues the DMA (TDM ignores EXEC)
    unsigned long long ga = (unsigned long long)(const void*)Wp;
    v4u g0;
    g0[0] = 1u;                                   // count=1, user mode
    g0[1] = 0u;                                   // lds_addr = 0
    g0[2] = (unsigned)(ga & 0xFFFFFFFFu);         // global_addr[31:0]
    g0[3] = (unsigned)((ga >> 32) & 0x1FFFFFFu)   // global_addr[56:32]
            | (2u << 30);                         // type = 2 ("image")
    v8i g1 = {};
    g1[0] = (int)(3u << 16);       // workgroup_mask=0, data_size=3 (8 bytes)
    g1[1] = (int)(8192u << 16);    // tensor_dim0[15:0] in [63:48]
    g1[2] = (int)(1u << 16);       // tensor_dim0[31:16]=0, tensor_dim1[15:0]=1
    g1[3] = (int)(8192u << 16);    // tensor_dim1[31:16]=0, tile_dim0=8192
    g1[4] = 1;                     // tile_dim1=1, tile_dim2=0
    g1[5] = 8192;                  // tensor_dim0_stride[31:0]
    g1[6] = 0;                     // stride hi, tensor_dim1_stride lo
    g1[7] = 0;
    v4i g2 = {};                   // dims 2/3 unused
    v4i g3 = {};
#if __clang_major__ >= 23
    v8i g4 = {};
    __builtin_amdgcn_tensor_load_to_lds(g0, g1, g2, g3, g4, 0);
#else
    __builtin_amdgcn_tensor_load_to_lds(g0, g1, g2, g3, 0);
#endif
    __builtin_amdgcn_s_wait_tensorcnt(0);
  }
  // The TDM builtin carries the LDS destination only as integer descriptor
  // bits, so LLVM cannot see that sW is written and would fold the B-matrix
  // loads to undef (observed: ds=0, wmma halved by CSE). Escape sW's address
  // into a may-write-memory asm so the optimizer must treat sW as modified.
  {
    _Float16* guard = &sW[0];
    asm volatile("" : : "v"(guard) : "memory");
  }
  __syncthreads();
#else
  {
    const int4* g = (const int4*)Wp;
    int4* s = (int4*)sW;
    for (int i = threadIdx.x; i < (32 * 2 * 32 * 16) / 8; i += blockDim.x) s[i] = g[i];
  }
  __syncthreads();
#endif

  const int lane = threadIdx.x & 31;
  const int wv   = threadIdx.x >> 5;
  const int wpb  = blockDim.x >> 5;
  const int half = lane >> 4;       // 0: K lo-halves, 1: K hi-halves
  const int lm   = lane & 15;
  const int nTiles = E / 16;

  for (int tile = blockIdx.x * wpb + wv; tile < nTiles; tile += gridDim.x * wpb) {
    const int base = tile * 16;
    // --- A-operand inputs: this lane feeds row M = lm of the 16x32 A tile ---
    const int eA = base + lm;
    const int sA = src[eA];
    const _Float16* ep = eah + (size_t)eA * DH;
    // A layout (contiguous K, matching packed B): lane<16 holds K {0..15},
    // lane>=16 holds K {16..31}.
    v8h ea0 = *(const v8h*)(ep + (half ? 16 : 0));
    v8h ea1 = *(const v8h*)(ep + (half ? 24 : 8));
    const _Float16* hp = hh + (size_t)sA * DH;
    v16h hA = *(const v16h*)hp;
    v16h hB = *(const v16h*)(hp + 16);

    v8f acc0 = {};  // columns 0..15
    v8f acc1 = {};  // columns 16..31
    #pragma unroll
    for (int i = 0; i < 32; ++i) {          // K chunk i <=> input feature i
      _Float16 hs = (i < 16) ? hA[i] : hB[i - 16];
      v16h a;
      #pragma unroll
      for (int j = 0; j < 8; ++j) { a[j] = ea0[j] * hs; a[j + 8] = ea1[j] * hs; }
      const v16h b0 = *(const v16h*)&sW[(size_t)(i * 2 + 0) * 512 + lane * 16];
      const v16h b1 = *(const v16h*)&sW[(size_t)(i * 2 + 1) * 512 + lane * 16];
      acc0 = __builtin_amdgcn_wmma_f32_16x16x32_f16(false, a, false, b0,
                                                    (short)0, acc0, false, false);
      acc1 = __builtin_amdgcn_wmma_f32_16x16x32_f16(false, a, false, b1,
                                                    (short)0, acc1, false, false);
    }

    // D layout: VGPR r, lane<16 -> (M=r, Nc=lane); lane>=16 -> (M=8+r, Nc=lane-16)
    const int n0 = lm, n1 = 16 + lm;
    #pragma unroll
    for (int r = 0; r < 8; ++r) {
      const int M = r + half * 8;
      const int e = base + M;
      const int s = src[e];
      const int d = dst[e];
      atomicAdd(&agg[(size_t)d * DH + n0], acc0[r] + hb[(size_t)s * DH + n0]);
      atomicAdd(&agg[(size_t)d * DH + n1], acc1[r] + hb[(size_t)s * DH + n1]);
    }
  }
}

// --------------------------------- launch ----------------------------------

extern "C" void kernel_launch(void* const* d_in, const int* in_sizes, int n_in,
                              void* d_out, int out_size, void* d_ws, size_t ws_size,
                              hipStream_t stream) {
  (void)in_sizes; (void)n_in; (void)out_size; (void)ws_size;

  const float* x         = (const float*)d_in[0];
  const float* edge_attr = (const float*)d_in[1];
  const int*   edge_idx  = (const int*)d_in[2];
  const int*   src = edge_idx;
  const int*   dst = edge_idx + E;
  const float* We_node = (const float*)d_in[3];  const float* be_node = (const float*)d_in[4];
  const float* We_edge = (const float*)d_in[5];  const float* be_edge = (const float*)d_in[6];
  const float* W1_nn   = (const float*)d_in[7];  const float* b1_nn   = (const float*)d_in[8];
  const float* root1   = (const float*)d_in[9];  const float* bias1   = (const float*)d_in[10];
  const float* Wl      = (const float*)d_in[11]; const float* bl      = (const float*)d_in[12];
  const float* Wr      = (const float*)d_in[13]; const float* br      = (const float*)d_in[14];
  const float* att     = (const float*)d_in[15]; const float* bias_gat= (const float*)d_in[16];
  const float* W2_nn   = (const float*)d_in[17]; const float* b2_nn   = (const float*)d_in[18];
  const float* root2   = (const float*)d_in[19]; const float* bias2   = (const float*)d_in[20];
  const float* Wlin    = (const float*)d_in[21]; const float* blin    = (const float*)d_in[22];
  float* out = (float*)d_out;

  // workspace carve (256B aligned)
  char* wsb = (char*)d_ws;
  size_t off = 0;
  auto carve = [&](size_t bytes) -> void* {
    void* p = wsb + off;
    off += (bytes + 255) & ~(size_t)255;
    return p;
  };
  float*    h0     = (float*)   carve((size_t)N * DH * 4);
  _Float16* h0h    = (_Float16*)carve((size_t)N * DH * 2);
  _Float16* eah    = (_Float16*)carve((size_t)E * DH * 2);
  _Float16* Wp1    = (_Float16*)carve(32 * 2 * 32 * 16 * 2);
  _Float16* Wp2    = (_Float16*)carve(32 * 2 * 32 * 16 * 2);
  float*    hb     = (float*)   carve((size_t)N * DH * 4);
  float*    agg    = (float*)   carve((size_t)N * DH * 4);
  float*    h1     = (float*)   carve((size_t)N * DH * 4);
  float*    xl     = (float*)   carve((size_t)N * DH * 4);
  float*    xr     = (float*)   carve((size_t)N * DH * 4);
  float*    logits = (float*)   carve((size_t)E2 * NH * 4);
  float*    mmax   = (float*)   carve((size_t)N * NH * 4);
  float*    denom  = (float*)   carve((size_t)N * NH * 4);
  float*    outgat = (float*)   carve((size_t)N * DH * 4);
  float*    h2     = (float*)   carve((size_t)N * DH * 4);
  _Float16* h2h    = (_Float16*)carve((size_t)N * DH * 2);
  float*    h3     = (float*)   carve((size_t)N * DH * 4);

  const int TB = 256;
  const int gN32  = (N * DH + TB - 1) / TB;
  const int gE32  = (E * DH + TB - 1) / TB;
  const int gPack = (32 * 2 * 32 * 16 + TB - 1) / TB;
  const int gE2H  = (E2 * NH + TB - 1) / TB;
  const int gNH   = (N * NH + TB - 1) / TB;
  const int gCls  = (N * NC + TB - 1) / TB;
  const int gConv = (E / 16 + 7) / 8;   // 8 waves (16-edge tiles) per block

  // encoders + weight packing
  k_encode_nodes<<<gN32, TB, 0, stream>>>(x, We_node, be_node, h0, h0h);
  k_encode_edges<<<gE32, TB, 0, stream>>>(edge_attr, We_edge, be_edge, eah);
  k_pack_w<<<gPack, TB, 0, stream>>>(W1_nn, Wp1);
  k_pack_w<<<gPack, TB, 0, stream>>>(W2_nn, Wp2);

  // ---- NNConv 1 ----
  k_hb<<<gN32, TB, 0, stream>>>(h0, b1_nn, hb);
  hipMemsetAsync(agg, 0, (size_t)N * DH * 4, stream);
  k_nnconv_wmma<<<gConv, TB, 0, stream>>>(eah, h0h, src, dst, Wp1, hb, agg);
  k_node_update<<<gN32, TB, 0, stream>>>(agg, h0, root1, bias1, h1, nullptr);

  // ---- GATv2 ----
  k_xlxr<<<gN32, TB, 0, stream>>>(h1, Wl, bl, Wr, br, xl, xr);
  k_init_minf<<<gNH, TB, 0, stream>>>((unsigned int*)mmax, N * NH);
  hipMemsetAsync(denom, 0, (size_t)N * NH * 4, stream);
  hipMemsetAsync(outgat, 0, (size_t)N * DH * 4, stream);
  k_gat_logits<<<gE2H, TB, 0, stream>>>(xl, xr, att, src, dst, logits, mmax);
  k_gat_exp<<<gE2H, TB, 0, stream>>>(logits, mmax, dst, denom);
  k_gat_agg<<<gE2H, TB, 0, stream>>>(logits, denom, xl, src, dst, outgat);
  k_gat_final<<<gN32, TB, 0, stream>>>(outgat, bias_gat, h2, h2h);

  // ---- NNConv 2 ----
  k_hb<<<gN32, TB, 0, stream>>>(h2, b2_nn, hb);
  hipMemsetAsync(agg, 0, (size_t)N * DH * 4, stream);
  k_nnconv_wmma<<<gConv, TB, 0, stream>>>(eah, h2h, src, dst, Wp2, hb, agg);
  k_node_update<<<gN32, TB, 0, stream>>>(agg, h2, root2, bias2, h3, nullptr);

  // ---- classifier ----
  k_classify<<<gCls, TB, 0, stream>>>(h3, Wlin, blin, out);
}
